// Generator_69088843924194
// MI455X (gfx1250) — matrix operand order repaired and checked by reading, not compiled, not run on hardware
//
#include <hip/hip_runtime.h>
#include <hip/hip_bf16.h>

typedef __attribute__((ext_vector_type(16))) _Float16 v16h;
typedef __attribute__((ext_vector_type(8)))  _Float16 v8h;
typedef __attribute__((ext_vector_type(8)))  float    v8f;

#define CH 128
#define MAX_ATOMS 768
#define RNN_LAYERS 4
#define NEG_SLOPE 0.2f

#define FRAG_HALVES (RNN_LAYERS * 2 * 8 * 4 * 512)          // 131072

// ---------------------------------------------------------------------------
// K1: length head.  3x (lrelu(h @ W + b)) then l = clip(|h@Wo + bo|,0,1),
// c = floor(l*768)+1.   Tiny: one block, plain VALU.
// ---------------------------------------------------------------------------
__global__ __launch_bounds__(128)
void len_head_kernel(const float* __restrict__ x,
                     const float* __restrict__ lenW,
                     const float* __restrict__ lenb,
                     const float* __restrict__ lenWo,
                     const float* __restrict__ lenbo,
                     float* __restrict__ l_out,   // d_out + 7680
                     int*   __restrict__ c_out) { // ws header
  __shared__ float h[CH];
  __shared__ float p[CH];
  const int t = threadIdx.x;
  h[t] = x[t];
  __syncthreads();
  for (int i = 0; i < 3; ++i) {
    float s = lenb[i * CH + t];
    for (int k = 0; k < CH; ++k) s += h[k] * lenW[i * CH * CH + k * CH + t];
    s = (s >= 0.f) ? s : NEG_SLOPE * s;
    __syncthreads();
    h[t] = s;
    __syncthreads();
  }
  p[t] = h[t] * lenWo[t];
  __syncthreads();
  if (t == 0) {
    float s = lenbo[0];
    for (int k = 0; k < CH; ++k) s += p[k];
    float l = fabsf(s);
    if (l > 1.f) l = 1.f;
    l_out[0] = l;
    c_out[0] = (int)floorf(l * (float)MAX_ATOMS) + 1;
  }
}

// ---------------------------------------------------------------------------
// K2: repack W_ih/W_hh (f32, [4][128 in][128 out]) into f16 WMMA B-fragment
// order.  B 32x16 f16 layout (wave32): half e of lane L holds
// B[K = kchunk*32 + (e<8?e:e+8) + (L<16?0:8)][N = nblk*16 + (L&15)].
// Fragment index f = ((layer*2 + mat)*8 + nblk)*4 + kchunk, 512 halves each.
// ---------------------------------------------------------------------------
__global__ __launch_bounds__(256)
void repack_kernel(const float* __restrict__ W_ih,
                   const float* __restrict__ W_hh,
                   _Float16* __restrict__ frag) {
  const int i = blockIdx.x * 256 + threadIdx.x;
  if (i >= FRAG_HALVES) return;
  const int f  = i >> 9;
  const int r  = i & 511;
  const int L  = r >> 4;
  const int e  = r & 15;
  const int kc = f & 3;
  const int nb = (f >> 2) & 7;
  const int m  = (f >> 5) & 1;
  const int li = f >> 6;
  const int K  = kc * 32 + ((e < 8) ? e : e + 8) + ((L < 16) ? 0 : 8);
  const int N  = nb * 16 + (L & 15);
  const float* W = m ? W_hh : W_ih;
  frag[i] = (_Float16)W[li * CH * CH + K * CH + N];
}

// ---------------------------------------------------------------------------
// K3: the sequential RNN.  One workgroup, 8 waves; wave w owns output block w
// (16 columns).  All weights register-resident (32 v16h = 256 VGPRs per wave,
// via gfx1250 VGPR-MSB).  Per layer: load cur A-frags (post-barrier) + use
// prefetched hid A-frags, 8 WMMAs in two 4-deep chains, write 16 outputs,
// prefetch next layer's hid frags (they predate the barrier), 1 barrier.
// Broadcast-A trick: every D row equals the matvec, so ALL lanes extract
// element 0 and write (duplicate lanes store identical values) - no exec
// divergence.  cur double-buffered on layer parity, hid on step parity.
// ---------------------------------------------------------------------------
__global__ __launch_bounds__(256)
void rnn_kernel(const float* __restrict__ x,
                const float* __restrict__ b_ih,
                const float* __restrict__ b_hh,
                const _Float16* __restrict__ frag,
                float* __restrict__ enc) {     // [768][128] f32
  __shared__ __align__(16) _Float16 curB[2 * CH];               // layer parity
  __shared__ __align__(16) _Float16 hidB[2 * RNN_LAYERS * CH];  // step parity

  const int t    = threadIdx.x;
  const int lane = t & 31;
  const int w    = t >> 5;                     // 0..7 : output block
  const int ksel = (lane < 16) ? 0 : 8;        // A layout lane-group K offset

  // Preamble: pull this wave's 32 weight fragments into registers (one-time).
  v16h wreg[RNN_LAYERS * 2 * 4];
#pragma unroll
  for (int li = 0; li < RNN_LAYERS; ++li)
#pragma unroll
    for (int m = 0; m < 2; ++m)
#pragma unroll
      for (int kc = 0; kc < 4; ++kc) {
        const int f = ((li * 2 + m) * 8 + w) * 4 + kc;
        wreg[(li * 2 + m) * 4 + kc] =
            *(const v16h*)(frag + f * 512 + lane * 16);
      }

  // Init state: input starts at zero, hid[0] = x, other layers zero.
  if (t < CH) {
    curB[t] = (_Float16)0.0f;
    curB[CH + t] = (_Float16)0.0f;
    for (int li = 0; li < RNN_LAYERS; ++li) {
      hidB[li * CH + t] = (li == 0) ? (_Float16)x[t] : (_Float16)0.0f;
      hidB[RNN_LAYERS * CH + li * CH + t] = (_Float16)0.0f;
    }
  }

  // Preload combined biases.
  float bias[RNN_LAYERS];
#pragma unroll
  for (int li = 0; li < RNN_LAYERS; ++li) {
    const int col = w * 16 + (lane & 15);
    bias[li] = b_ih[li * CH + col] + b_hh[li * CH + col];
  }
  __syncthreads();

  // Prefetch hid A-fragments for (step 0, layer 0).
  v16h ah[4];
#pragma unroll
  for (int kc = 0; kc < 4; ++kc) {
    v8h lo = *(const v8h*)&hidB[kc * 32 + ksel];
    v8h hi = *(const v8h*)&hidB[kc * 32 + 16 + ksel];
#pragma unroll
    for (int e = 0; e < 8; ++e) { ah[kc][e] = lo[e]; ah[kc][e + 8] = hi[e]; }
  }

  for (int step = 0; step < MAX_ATOMS; ++step) {
    const int p = step & 1;
#pragma unroll
    for (int li = 0; li < RNN_LAYERS; ++li) {
      const _Float16* curR = curB + (li & 1) * CH;
      _Float16*       curW = curB + ((li + 1) & 1) * CH;
      _Float16*       hidW = hidB + (1 - p) * (RNN_LAYERS * CH) + li * CH;

      // cur A-fragments (depend on previous layer's writes -> post-barrier).
      v16h ai[4];
#pragma unroll
      for (int kc = 0; kc < 4; ++kc) {
        v8h lo = *(const v8h*)&curR[kc * 32 + ksel];
        v8h hi = *(const v8h*)&curR[kc * 32 + 16 + ksel];
#pragma unroll
        for (int e = 0; e < 8; ++e) { ai[kc][e] = lo[e]; ai[kc][e + 8] = hi[e]; }
      }

      // Two independent 4-deep WMMA chains; weights already in registers.
      v8f acc_i, acc_h;
#pragma unroll
      for (int j = 0; j < 8; ++j) { acc_i[j] = 0.0f; acc_h[j] = 0.0f; }
#pragma unroll
      for (int kc = 0; kc < 4; ++kc) {
        acc_i = __builtin_amdgcn_wmma_f32_16x16x32_f16(
            false, ai[kc], false, wreg[(li * 2 + 0) * 4 + kc],
            (short)0, acc_i, false, false);
        acc_h = __builtin_amdgcn_wmma_f32_16x16x32_f16(
            false, ah[kc], false, wreg[(li * 2 + 1) * 4 + kc],
            (short)0, acc_h, false, false);
      }

      // Extraction: all lanes (lanes 16-31 duplicate lanes 0-15's values).
      {
        const int col = w * 16 + (lane & 15);
        float v = acc_i[0] + acc_h[0] + bias[li];
        v = fmaxf(v, 0.0f);                   // relu
        _Float16 hv = (_Float16)v;
        hidW[col] = hv;                       // state for next step
        curW[col] = hv;                       // input for next layer / step
        if (li == RNN_LAYERS - 1) enc[step * CH + col] = v;
      }

      // Prefetch next layer's hid fragments: that buffer predates the
      // barrier (written last step, or in layer 0 of this step for li==3),
      // so these loads legally overlap the barrier wait.
      {
        const _Float16* hN = (li < RNN_LAYERS - 1)
            ? (hidB + p * (RNN_LAYERS * CH) + (li + 1) * CH)          // this step
            : (hidB + (1 - p) * (RNN_LAYERS * CH));                   // next step, layer 0
#pragma unroll
        for (int kc = 0; kc < 4; ++kc) {
          v8h lo = *(const v8h*)&hN[kc * 32 + ksel];
          v8h hi = *(const v8h*)&hN[kc * 32 + 16 + ksel];
#pragma unroll
          for (int e = 0; e < 8; ++e) { ah[kc][e] = lo[e]; ah[kc][e + 8] = hi[e]; }
        }
      }
      __syncthreads();                        // writes visible to next layer
    }
  }
}

// ---------------------------------------------------------------------------
// K4: residual heads.  res = lrelu(e + e@rW + rb); out = res @ W.
// Per block: one 16-row tile of enc; the 128x128 residual GEMM on WMMA,
// the tiny 128->{8,1,1} projections on VALU.  Mask rows >= c with zeros.
// ---------------------------------------------------------------------------
__global__ __launch_bounds__(128)
void heads_kernel(const float* __restrict__ enc, const int* __restrict__ cptr,
                  const float* __restrict__ atom_rW, const float* __restrict__ atom_rb,
                  const float* __restrict__ atom_W,
                  const float* __restrict__ pos_rW,  const float* __restrict__ pos_rb,
                  const float* __restrict__ pos_W,
                  const float* __restrict__ mag_rW,  const float* __restrict__ mag_rb,
                  const float* __restrict__ mag_W,
                  float* __restrict__ out) {
  __shared__ float res[16][CH + 4];
  const int t    = threadIdx.x;
  const int lane = t & 31;
  const int w    = t >> 5;
  const int M    = lane & 15;
  const int ksel = (lane < 16) ? 0 : 8;
  const int R    = blockIdx.x * 16;
  const int cval = *cptr;

  // A fragments: 16 real rows of enc tile, converted f32 -> f16.
  v16h afr[4];
  const float* er = enc + (R + M) * CH;
#pragma unroll
  for (int kc = 0; kc < 4; ++kc) {
#pragma unroll
    for (int e = 0; e < 8; ++e) {
      afr[kc][e]     = (_Float16)er[kc * 32 + ksel + e];
      afr[kc][e + 8] = (_Float16)er[kc * 32 + 16 + ksel + e];
    }
  }

  const float* rWs[3] = {atom_rW, pos_rW, mag_rW};
  const float* rbs[3] = {atom_rb, pos_rb, mag_rb};

  for (int hd = 0; hd < 3; ++hd) {
    const float* rW = rWs[hd];
    const float* rb = rbs[hd];
#pragma unroll
    for (int nb = 0; nb < 2; ++nb) {
      const int NB = w * 2 + nb;
      v8f acc;
#pragma unroll
      for (int j = 0; j < 8; ++j) acc[j] = 0.0f;
#pragma unroll
      for (int kc = 0; kc < 4; ++kc) {
        v16h b;
#pragma unroll
        for (int e = 0; e < 16; ++e) {
          const int K = kc * 32 + ((e < 8) ? e : e + 8) + ((lane < 16) ? 0 : 8);
          const int N = NB * 16 + M;
          b[e] = (_Float16)rW[K * CH + N];
        }
        acc = __builtin_amdgcn_wmma_f32_16x16x32_f16(
            false, afr[kc], false, b, (short)0, acc, false, false);
      }
      // D layout: elem j of lane L holds D[j + (L<16?0:8)][(L&15)] of tile.
#pragma unroll
      for (int j = 0; j < 8; ++j) {
        const int Mr  = j + ((lane < 16) ? 0 : 8);
        const int col = NB * 16 + M;
        float v = enc[(R + Mr) * CH + col] + acc[j] + rb[col];
        v = (v >= 0.f) ? v : NEG_SLOPE * v;   // lrelu
        res[Mr][col] = v;
      }
    }
    __syncthreads();

    if (hd == 0) {                 // atoms: 16 rows x 8 cols = 128 dots
      const int row = t >> 3, cc = t & 7;
      float s = 0.f;
      for (int k = 0; k < CH; ++k) s += res[row][k] * atom_W[k * 8 + cc];
      out[(R + row) * 10 + cc] = ((R + row) < cval) ? s : 0.f;
    } else if (t < 16) {           // pos / mags: 16 dots
      const float* W = (hd == 1) ? pos_W : mag_W;
      float s = 0.f;
      for (int k = 0; k < CH; ++k) s += res[t][k] * W[k];
      out[(R + t) * 10 + ((hd == 1) ? 8 : 9)] = ((R + t) < cval) ? s : 0.f;
    }
    __syncthreads();
  }
}

// ---------------------------------------------------------------------------
// Host-side launcher.
// Workspace layout: [0,256)              int c
//                   [256, 256+393216)    enc f32 [768][128]
//                   [+262144)            repacked RNN weights (f16)
// ---------------------------------------------------------------------------
extern "C" void kernel_launch(void* const* d_in, const int* in_sizes, int n_in,
                              void* d_out, int out_size, void* d_ws, size_t ws_size,
                              hipStream_t stream) {
  const float* x       = (const float*)d_in[0];
  const float* len_W   = (const float*)d_in[1];
  const float* len_b   = (const float*)d_in[2];
  const float* len_Wo  = (const float*)d_in[3];
  const float* len_bo  = (const float*)d_in[4];
  const float* W_ih    = (const float*)d_in[5];
  const float* W_hh    = (const float*)d_in[6];
  const float* b_ih    = (const float*)d_in[7];
  const float* b_hh    = (const float*)d_in[8];
  const float* atom_rW = (const float*)d_in[9];
  const float* atom_rb = (const float*)d_in[10];
  const float* atom_W  = (const float*)d_in[11];
  const float* pos_rW  = (const float*)d_in[12];
  const float* pos_rb  = (const float*)d_in[13];
  const float* pos_W   = (const float*)d_in[14];
  const float* mag_rW  = (const float*)d_in[15];
  const float* mag_rb  = (const float*)d_in[16];
  const float* mag_W   = (const float*)d_in[17];

  float* out = (float*)d_out;                         // 7680 recon + 1 l
  char* ws = (char*)d_ws;
  int*      c_ws = (int*)ws;
  float*    enc  = (float*)(ws + 256);
  _Float16* frag = (_Float16*)(ws + 256 + MAX_ATOMS * CH * sizeof(float));

  len_head_kernel<<<1, 128, 0, stream>>>(x, len_W, len_b, len_Wo, len_bo,
                                         out + MAX_ATOMS * 10, c_ws);
  repack_kernel<<<512, 256, 0, stream>>>(W_ih, W_hh, frag);
  rnn_kernel<<<1, 256, 0, stream>>>(x, b_ih, b_hh, frag, enc);
  heads_kernel<<<MAX_ATOMS / 16, 128, 0, stream>>>(enc, c_ws,
                                                   atom_rW, atom_rb, atom_W,
                                                   pos_rW, pos_rb, pos_W,
                                                   mag_rW, mag_rb, mag_W, out);
}
